// TreeStructuredRCNNLSTM_274877907231
// MI455X (gfx1250) — matrix operand order repaired and checked by reading, not compiled, not run on hardware
//
#include <hip/hip_runtime.h>
#include <hip/hip_bf16.h>
#include <cstdint>

#define DEVFN __device__ __forceinline__

// ---- problem constants ----
#define BB 128
#define SS 512
#define BS 65536     // B*S rows
#define EE 300
#define EP 320       // E padded to 10 k-steps of 32
#define FP 608       // feats width 600 padded to 19*32
#define HP 160       // H=150 padded to 160 (5 k-steps of 32, 10 n-tiles)
#define GP 768       // 5H=750 padded to 48 n-tiles

typedef __attribute__((ext_vector_type(16))) __bf16 v16bf;
typedef __attribute__((ext_vector_type(8)))  float  v8f;

union FragU { v16bf v; float4 f4[2]; };

DEVFN __bf16 f2bf(float f) {
  union { float f; unsigned u; } a; a.f = f;
  unsigned r = (a.u + 0x7FFFu + ((a.u >> 16) & 1u)) >> 16;
  union { unsigned short s; __bf16 b; } o; o.s = (unsigned short)r; return o.b;
}
DEVFN float bf2f(__bf16 b) {
  union { unsigned short s; __bf16 b; } i; i.b = b;
  union { unsigned u; float f; } o; o.u = ((unsigned)i.s) << 16; return o.f;
}
DEVFN float sigf(float x) { return 1.f / (1.f + __expf(-x)); }

DEVFN v8f wmma_bf16(v16bf a, v16bf b, v8f c) {
  return __builtin_amdgcn_wmma_f32_16x16x32_bf16(false, a, false, b, (short)0, c,
                                                 false, false);
}

// Async global -> LDS copy (CDNA5, ASYNCcnt-tracked). lds is a pointer into a
// __shared__ array (low 32 bits of the generic pointer are the LDS offset).
DEVFN void async_ldg_b128(void* lds, const void* g) {
  unsigned lds_off = (unsigned)(uintptr_t)lds;
  asm volatile("global_load_async_to_lds_b128 %0, %1, off"
               :: "v"(lds_off), "v"(g) : "memory");
}
DEVFN void wait_async0() {
  asm volatile("s_wait_asynccnt 0" ::: "memory");
}

// A fragment (16x32 bf16) from LDS tile, row-major, ld elems.
// Lanes 0-15: row m, K 0..15; lanes 16-31: K 16..31.
DEVFN v16bf load_a_frag(const __bf16* Alds, int lane, int mbase, int ld, int kbase) {
  int m  = mbase + (lane & 15);
  int kh = (lane >> 4) << 4;
  const __bf16* p = Alds + m * ld + kbase + kh;
  FragU fr;
  fr.f4[0] = *(const float4*)(p);
  fr.f4[1] = *(const float4*)(p + 8);
  return fr.v;
}

// B fragment (32x16 bf16) straight from global, weights stored N-major [n][k].
// Lanes 0-15: K 0..15, lanes 16-31: K 16..31; n = nbase + (lane&15).
DEVFN v16bf load_b_glob(const __bf16* Bt, int lane, int nbase, int kbase, int ld) {
  const __bf16* p = Bt + (size_t)(nbase + (lane & 15)) * ld + kbase + ((lane >> 4) << 4);
  FragU fr;
  fr.f4[0] = ((const float4*)p)[0];
  fr.f4[1] = ((const float4*)p)[1];
  return fr.v;
}

// ---------------- weight repack (fp32 -> padded N-major bf16) ----------------
__global__ void k_prep(const float* w3, const float* w4, const float* w5,
                       const float* Wp, const float* Wx, const float* UL, const float* UR,
                       __bf16* WoT, __bf16* WpT, __bf16* WxT, __bf16* ULT, __bf16* URT) {
  const int N0 = 5 * 320 * 320;       // WoT
  const int N1 = HP * FP;             // WpT
  const int N2 = GP * HP;             // WxT / ULT / URT each
  const int total = N0 + N1 + 3 * N2;
  int stride = gridDim.x * blockDim.x;
  for (int idx = blockIdx.x * blockDim.x + threadIdx.x; idx < total; idx += stride) {
    float v = 0.f;
    if (idx < N0) {
      int o = idx / 102400; int r = idx - o * 102400;
      int n = r / 320, k = r - (r / 320) * 320;
      int off = o - 2;
      if (k < EE) {
        if (n < 100)      { int d = off + 1; if (d >= 0 && d < 3) v = w3[n * 900 + k * 3 + d]; }
        else if (n < 200) { int f = n - 100; int d = off + 1; if (d >= 0 && d < 4) v = w4[f * 1200 + k * 4 + d]; }
        else if (n < 300) { int f = n - 200; int d = off + 2; v = w5[f * 1500 + k * 5 + d]; }
      }
      WoT[idx] = f2bf(v);
    } else if (idx < N0 + N1) {
      int r = idx - N0;
      int n = r / FP, k = r - n * FP;
      if (n < 150 && k < 600) v = Wp[n * 600 + k];
      WpT[r] = f2bf(v);
    } else {
      int r = idx - (N0 + N1);
      int which = r / N2; int r2 = r - which * N2;
      int n = r2 / HP, k = r2 - n * HP;
      const float* src = (which == 0) ? Wx : (which == 1) ? UL : UR;
      if (n < 750 && k < 150) v = src[n * 150 + k];
      __bf16* dst = (which == 0) ? WxT : (which == 1) ? ULT : URT;
      dst[r2] = f2bf(v);
    }
  }
}

// ---------------- embedding gather -> bf16 e_buf + feats ----------------
__global__ void k_gather(const int* x, const float* emb, __bf16* e_buf, __bf16* feats) {
  int stride = gridDim.x * blockDim.x;
  for (int idx = blockIdx.x * blockDim.x + threadIdx.x; idx < BS * FP; idx += stride) {
    int r = idx / FP, col = idx - r * FP;
    float v = 0.f;
    if (col < EE) v = emb[(size_t)x[r] * EE + col];
    __bf16 bv = f2bf(v);
    feats[(size_t)r * FP + col] = bv;       // cols >= 600 stay zero
    if (col < EP) e_buf[(size_t)r * EP + col] = bv;
  }
}

// ---------------- conv as 5 shifted GEMMs: feats[:,300:600] ----------------
// Flattened 50-step (offset x k-chunk) loop, double-buffered async A staging.
__global__ __launch_bounds__(256) void k_conv(const __bf16* __restrict__ e_buf,
                                              const __bf16* __restrict__ WoT,
                                              const float* b3, const float* b4,
                                              const float* b5, __bf16* feats) {
  __shared__ __align__(16) __bf16 Alds[2][128 * 40];
  int tid = threadIdx.x, lane = tid & 31, wave = tid >> 5;
  int r0 = blockIdx.x * 128, n0 = blockIdx.y * 64;
  v8f acc[4] = {};

  auto issue_tile = [&](int t, int buf) {
    int off = t / 10 - 2;
    int kb  = (t - (t / 10) * 10) * 32;
    for (int c = tid; c < 512; c += 256) {            // 128x32 tile, 16B chunks
      int m = c >> 2, kc = (c & 3) << 3;
      int rg = r0 + m;
      __bf16* dst = &Alds[buf][m * 40 + kc];
      if ((unsigned)((rg & (SS - 1)) + off) < (unsigned)SS) {
        async_ldg_b128(dst, e_buf + (size_t)(rg + off) * EP + kb + kc);
      } else {
        *(float4*)dst = make_float4(0.f, 0.f, 0.f, 0.f);
      }
    }
  };

  issue_tile(0, 0);
  for (int t = 0; t < 50; ++t) {
    int cur = t & 1;
    wait_async0();
    __syncthreads();                                  // tile t visible to all waves
    if (t + 1 < 50) issue_tile(t + 1, cur ^ 1);       // overlap with compute below
    const __bf16* Wb = WoT + (size_t)(t / 10) * (320 * 320);
    int kb = (t - (t / 10) * 10) * 32;
    v16bf a = load_a_frag(Alds[cur], lane, wave * 16, 40, 0);
    v16bf bfr[4];
#pragma unroll
    for (int nt = 0; nt < 4; ++nt)
      bfr[nt] = load_b_glob(Wb, lane, n0 + nt * 16, kb, 320);
#pragma unroll
    for (int nt = 0; nt < 4; ++nt)
      acc[nt] = wmma_bf16(a, bfr[nt], acc[nt]);
  }
#pragma unroll
  for (int nt = 0; nt < 4; ++nt) {
    int n = n0 + nt * 16 + (lane & 15);
    if (n < 300) {
      float bias = (n < 100) ? b3[n] : (n < 200) ? b4[n - 100] : b5[n - 200];
#pragma unroll
      for (int j = 0; j < 8; ++j) {
        int r = r0 + wave * 16 + ((lane >> 4) << 3) + j;
        float vv = acc[nt][j] + bias;
        vv = vv > 0.f ? vv : 0.f;
        feats[(size_t)r * FP + 300 + n] = f2bf(vv);
      }
    }
  }
}

// ---------------- leaf = feats @ Wp^T + bp ----------------
__global__ __launch_bounds__(256) void k_leaf(const __bf16* __restrict__ feats,
                                              const __bf16* __restrict__ WpT,
                                              const float* bp, __bf16* leaf) {
  __shared__ __align__(16) __bf16 Alds[2][128 * 40];
  int tid = threadIdx.x, lane = tid & 31, wave = tid >> 5;
  int r0 = blockIdx.x * 128;
  v8f acc[10] = {};

  auto issue_tile = [&](int kk, int buf) {
    for (int c = tid; c < 512; c += 256) {
      int m = c >> 2, kc = (c & 3) << 3;
      async_ldg_b128(&Alds[buf][m * 40 + kc],
                     feats + (size_t)(r0 + m) * FP + kk * 32 + kc);
    }
  };

  issue_tile(0, 0);
  for (int kk = 0; kk < 19; ++kk) {
    int cur = kk & 1;
    wait_async0();
    __syncthreads();
    if (kk + 1 < 19) issue_tile(kk + 1, cur ^ 1);
    v16bf a = load_a_frag(Alds[cur], lane, wave * 16, 40, 0);
    v16bf bfr[5];
#pragma unroll
    for (int nt = 0; nt < 5; ++nt)
      bfr[nt] = load_b_glob(WpT, lane, nt * 16, kk * 32, FP);
#pragma unroll
    for (int nt = 0; nt < 5; ++nt)
      acc[nt] = wmma_bf16(a, bfr[nt], acc[nt]);
#pragma unroll
    for (int nt = 0; nt < 5; ++nt)
      bfr[nt] = load_b_glob(WpT, lane, (nt + 5) * 16, kk * 32, FP);
#pragma unroll
    for (int nt = 0; nt < 5; ++nt)
      acc[nt + 5] = wmma_bf16(a, bfr[nt], acc[nt + 5]);
  }
#pragma unroll
  for (int nt = 0; nt < 10; ++nt) {
    int n = nt * 16 + (lane & 15);
    float bias = (n < 150) ? bp[n] : 0.f;
#pragma unroll
    for (int j = 0; j < 8; ++j) {
      int r = r0 + wave * 16 + ((lane >> 4) << 3) + j;
      leaf[(size_t)r * HP + n] = f2bf(acc[nt][j] + bias);
    }
  }
}

// ---------------- gates = leaf @ Wx^T + bx, then leaf LSTM cell ----------------
__global__ __launch_bounds__(256) void k_gates_leaf(const __bf16* __restrict__ leaf,
                                                    const __bf16* __restrict__ WxT,
                                                    const float* bx,
                                                    __bf16* h_out, float* c_out) {
  __shared__ __align__(16) __bf16 Alds[16 * 168];
  __shared__ float G[16 * GP];
  int tid = threadIdx.x, lane = tid & 31, wave = tid >> 5;
  int r0 = blockIdx.x * 16;
  for (int c = tid; c < 320; c += 256) {               // 16x160 bf16, 16B chunks
    int m = c / 20, kc = (c % 20) << 3;
    async_ldg_b128(Alds + m * 168 + kc, leaf + (size_t)(r0 + m) * HP + kc);
  }
  wait_async0();
  __syncthreads();
  v8f acc[6] = {};
  for (int kk = 0; kk < 5; ++kk) {
    v16bf a = load_a_frag(Alds, lane, 0, 168, kk * 32);
    v16bf bfr[6];
#pragma unroll
    for (int i = 0; i < 6; ++i)
      bfr[i] = load_b_glob(WxT, lane, (wave * 6 + i) * 16, kk * 32, HP);
#pragma unroll
    for (int i = 0; i < 6; ++i)
      acc[i] = wmma_bf16(a, bfr[i], acc[i]);
  }
#pragma unroll
  for (int i = 0; i < 6; ++i) {
    int n = (wave * 6 + i) * 16 + (lane & 15);
    float bias = (n < 750) ? bx[n] : 0.f;
#pragma unroll
    for (int j = 0; j < 8; ++j) {
      int m = ((lane >> 4) << 3) + j;
      G[m * GP + n] = acc[i][j] + bias;
    }
  }
  __syncthreads();
  for (int idx = tid; idx < 16 * HP; idx += 256) {
    int m = idx / HP, hc = idx - (idx / HP) * HP;
    int r = r0 + m;
    float hv = 0.f, cv = 0.f;
    if (hc < 150) {
      const float* g = G + m * GP;
      float ig = sigf(g[hc]);
      float og = sigf(g[450 + hc]);
      float u  = tanhf(g[600 + hc]);
      cv = ig * u;
      hv = og * tanhf(cv);
    }
    h_out[(size_t)r * HP + hc] = f2bf(hv);
    c_out[(size_t)r * HP + hc] = cv;
  }
}

// ---------------- one tree level: gates = lh@UL^T + rh@UR^T + bx, cell ----------------
__global__ __launch_bounds__(256) void k_tree(const __bf16* __restrict__ h_in,
                                              const float* __restrict__ c_in,
                                              __bf16* h_out, float* c_out,
                                              const __bf16* __restrict__ ULT,
                                              const __bf16* __restrict__ URT,
                                              const float* bx, int n_in, int lh) {
  __shared__ __align__(16) __bf16 Ae[16 * 168];
  __shared__ __align__(16) __bf16 Ao[16 * 168];
  __shared__ float G[16 * GP];
  int tid = threadIdx.x, lane = tid & 31, wave = tid >> 5;
  int node0 = blockIdx.x * 16;
  int mask = (n_in >> 1) - 1;
  for (int c = tid; c < 640; c += 256) {               // even+odd 16x160 tiles
    int eo = (c >= 320) ? 1 : 0;
    int c2 = c - eo * 320;
    int m = c2 / 20, kc = (c2 % 20) << 3;
    int g = node0 + m;
    int r = ((g >> lh) << (lh + 1)) + ((g & mask) << 1) + eo;   // b*n_in + 2j + eo
    __bf16* dst = (eo ? Ao : Ae) + m * 168 + kc;
    async_ldg_b128(dst, h_in + (size_t)r * HP + kc);
  }
  wait_async0();
  __syncthreads();
  v8f acc[6] = {};
  for (int kk = 0; kk < 5; ++kk) {
    v16bf aE = load_a_frag(Ae, lane, 0, 168, kk * 32);
    v16bf aO = load_a_frag(Ao, lane, 0, 168, kk * 32);
#pragma unroll
    for (int i = 0; i < 6; ++i) {
      int nb = (wave * 6 + i) * 16;
      v16bf bL = load_b_glob(ULT, lane, nb, kk * 32, HP);
      v16bf bR = load_b_glob(URT, lane, nb, kk * 32, HP);
      acc[i] = wmma_bf16(aE, bL, acc[i]);
      acc[i] = wmma_bf16(aO, bR, acc[i]);
    }
  }
#pragma unroll
  for (int i = 0; i < 6; ++i) {
    int n = (wave * 6 + i) * 16 + (lane & 15);
    float bias = (n < 750) ? bx[n] : 0.f;
#pragma unroll
    for (int j = 0; j < 8; ++j) {
      int m = ((lane >> 4) << 3) + j;
      G[m * GP + n] = acc[i][j] + bias;
    }
  }
  __syncthreads();
  for (int idx = tid; idx < 16 * HP; idx += 256) {
    int m = idx / HP, hc = idx - (idx / HP) * HP;
    int g = node0 + m;
    int rE = ((g >> lh) << (lh + 1)) + ((g & mask) << 1);
    float hv = 0.f, cv = 0.f;
    if (hc < 150) {
      const float* gg = G + m * GP;
      float ig = sigf(gg[hc]);
      float fL = sigf(gg[150 + hc]);
      float fR = sigf(gg[300 + hc]);
      float og = sigf(gg[450 + hc]);
      float u  = tanhf(gg[600 + hc]);
      cv = ig * u + fL * c_in[(size_t)rE * HP + hc] + fR * c_in[(size_t)(rE + 1) * HP + hc];
      hv = og * tanhf(cv);
    }
    h_out[(size_t)g * HP + hc] = f2bf(hv);
    c_out[(size_t)g * HP + hc] = cv;
  }
}

// ---------------- tiny MLP head ----------------
__global__ void k_final(const __bf16* __restrict__ hroot, const float* W1, const float* b1,
                        const float* W2, const float* b2, float* out) {
  int t = threadIdx.x;
  if (t >= BB) return;
  float z[75];
  for (int j = 0; j < 75; ++j) {
    float s = b1[j];
    for (int k = 0; k < 150; ++k) s += bf2f(hroot[(size_t)t * HP + k]) * W1[j * 150 + k];
    z[j] = s > 0.f ? s : 0.f;
  }
  for (int q = 0; q < 3; ++q) {
    float s = b2[q];
    for (int j = 0; j < 75; ++j) s += z[j] * W2[q * 75 + j];
    out[t * 3 + q] = s;
  }
}

extern "C" void kernel_launch(void* const* d_in, const int* in_sizes, int n_in,
                              void* d_out, int out_size, void* d_ws, size_t ws_size,
                              hipStream_t stream) {
  (void)in_sizes; (void)n_in; (void)out_size; (void)ws_size;
  const int*   x   = (const int*)d_in[0];
  const float* emb = (const float*)d_in[1];
  const float* w3  = (const float*)d_in[2];
  const float* b3  = (const float*)d_in[3];
  const float* w4  = (const float*)d_in[4];
  const float* b4  = (const float*)d_in[5];
  const float* w5  = (const float*)d_in[6];
  const float* b5  = (const float*)d_in[7];
  const float* Wp  = (const float*)d_in[8];
  const float* bp  = (const float*)d_in[9];
  const float* Wx  = (const float*)d_in[10];
  const float* bx  = (const float*)d_in[11];
  const float* UL  = (const float*)d_in[12];
  const float* UR  = (const float*)d_in[13];
  const float* W1  = (const float*)d_in[14];
  const float* b1  = (const float*)d_in[15];
  const float* W2  = (const float*)d_in[16];
  const float* b2  = (const float*)d_in[17];
  float* out = (float*)d_out;

  uintptr_t base = (uintptr_t)d_ws;
  auto take = [&](size_t bytes) -> void* {
    base = (base + 255) & ~(uintptr_t)255;
    void* p = (void*)base;
    base += bytes;
    return p;
  };
  __bf16* e_buf = (__bf16*)take((size_t)BS * EP * 2);
  __bf16* feats = (__bf16*)take((size_t)BS * FP * 2);
  __bf16* leafb = (__bf16*)take((size_t)BS * HP * 2);
  __bf16* h0    = (__bf16*)take((size_t)BS * HP * 2);
  __bf16* h1    = (__bf16*)take((size_t)BS * HP * 2);
  float*  c0    = (float*)take((size_t)BS * HP * 4);
  float*  c1    = (float*)take((size_t)BS * HP * 4);
  __bf16* WoT   = (__bf16*)take((size_t)5 * 320 * 320 * 2);
  __bf16* WpT   = (__bf16*)take((size_t)HP * FP * 2);
  __bf16* WxT   = (__bf16*)take((size_t)GP * HP * 2);
  __bf16* ULT   = (__bf16*)take((size_t)GP * HP * 2);
  __bf16* URT   = (__bf16*)take((size_t)GP * HP * 2);

  k_prep<<<512, 256, 0, stream>>>(w3, w4, w5, Wp, Wx, UL, UR, WoT, WpT, WxT, ULT, URT);
  k_gather<<<4096, 256, 0, stream>>>(x, emb, e_buf, feats);
  k_conv<<<dim3(BS / 128, 5), 256, 0, stream>>>(e_buf, WoT, b3, b4, b5, feats);
  k_leaf<<<BS / 128, 256, 0, stream>>>(feats, WpT, bp, leafb);
  k_gates_leaf<<<BS / 16, 256, 0, stream>>>(leafb, WxT, bx, h0, c0);

  const __bf16* hi = h0; const float* ci = c0;
  __bf16* ho = h1; float* co = c1;
  int n = SS;
  while (n > 1) {
    int hn = n >> 1;
    int lh = 0; { int t = hn; while (t > 1) { t >>= 1; ++lh; } }
    k_tree<<<8 * hn, 256, 0, stream>>>(hi, ci, ho, co, ULT, URT, bx, n, lh);
    const __bf16* th = hi; hi = ho; ho = (__bf16*)th;
    const float*  tc = ci; ci = co; co = (float*)tc;
    n = hn;
  }
  k_final<<<1, 128, 0, stream>>>(hi, W1, b1, W2, b2, out);
}
